// GTGModule_66864050864582
// MI455X (gfx1250) — matrix operand order repaired and checked by reading, not compiled, not run on hardware
//
#include <hip/hip_runtime.h>
#include <hip/hip_bf16.h>
#include <cstdint>

// ---------------------------------------------------------------------------
// Types for CDNA5 WMMA / TDM
// ---------------------------------------------------------------------------
typedef __attribute__((ext_vector_type(16))) __bf16 v16bf;
typedef __attribute__((ext_vector_type(8)))  __bf16 v8bf;
typedef __attribute__((ext_vector_type(8)))  float  v8f;
typedef __attribute__((ext_vector_type(4))) unsigned int v4u;
typedef __attribute__((ext_vector_type(4))) int  v4i_t;
typedef __attribute__((ext_vector_type(8))) int  v8i_t;

__device__ inline __bf16 f2bf(float f) {
    unsigned u = __builtin_bit_cast(unsigned, f);
    unsigned r = u + 0x7FFFu + ((u >> 16) & 1u);   // round-to-nearest-even
    unsigned short h = (unsigned short)(r >> 16);
    return __builtin_bit_cast(__bf16, h);
}

#define BSZ   4096
#define NCLS  10
#define TITER 30

// ---------------------------------------------------------------------------
// Tensor Data Mover: DMA one 128-row x 32-col bf16 tile (row-major, row
// stride K elements) from global into LDS. 2D descriptor per ISA 08 §8.3/8.4:
//   group0: count=1 | lds_addr | global_addr(tile start) | type=2
//   group1: data_size=1 (2B) | tensor_dim0=K | tensor_dim1=4096 |
//           tile_dim0=32 | tile_dim1=128 | tensor_dim0_stride=K
//   groups 2/3: zero (2D tile)
// Tracked with TENSORcnt (in-order), waited via s_wait_tensorcnt.
// ---------------------------------------------------------------------------
__device__ inline void tdm_load_tile(unsigned ldsAddr, const __bf16* gbase,
                                     int row0, int K, int kc) {
    unsigned long long ga =
        (unsigned long long)(uintptr_t)(gbase + (size_t)row0 * K + kc);
    v4u g0;
    g0[0] = 1u;                                        // count=1, user mode
    g0[1] = ldsAddr;                                   // LDS byte address
    g0[2] = (unsigned)(ga & 0xFFFFFFFFu);              // global_addr[31:0]
    g0[3] = (unsigned)((ga >> 32) & 0x01FFFFFFu)       // global_addr[56:32]
          | (2u << 30);                                // type=2 (image)
    v8i_t g1;
    g1[0] = (int)(1u << 16);                           // data_size=1 -> 2 bytes
    g1[1] = (int)(((unsigned)K & 0xFFFFu) << 16);      // tensor_dim0[15:0]
    g1[2] = (int)((((unsigned)K >> 16) & 0xFFFFu)      // tensor_dim0[31:16]
          | (4096u << 16));                            // tensor_dim1[15:0]
    g1[3] = (int)(32u << 16);                          // tile_dim0=32
    g1[4] = 128;                                       // tile_dim1=128
    g1[5] = K;                                         // dim0_stride[31:0]
    g1[6] = 0;
    g1[7] = 0;
    v4i_t gz = {0, 0, 0, 0};
#if __clang_major__ >= 23
    v8i_t gz8 = {0, 0, 0, 0, 0, 0, 0, 0};
    __builtin_amdgcn_tensor_load_to_lds(g0, g1, gz, gz, gz8, 0);
#else
    __builtin_amdgcn_tensor_load_to_lds(g0, g1, gz, gz, 0);
#endif
}

// ---------------------------------------------------------------------------
// Spatial average pool: feat [B,C,s,s] -> pooled [B,C]
// ---------------------------------------------------------------------------
__global__ void pool_kernel(const float* __restrict__ f, float* __restrict__ pooled,
                            int C, int S2) {
    int idx = blockIdx.x * 256 + threadIdx.x;
    if (idx >= BSZ * C) return;
    const float* p = f + (size_t)idx * S2;
    float s = 0.f;
    for (int i = 0; i < S2; ++i) s += p[i];
    pooled[idx] = s * (1.0f / (float)S2);
}

// ---------------------------------------------------------------------------
// Generic small GEMM: Y[m, colOff+n] = relu?(X[m,:K] @ W[:K, n] + bias[n])
// ---------------------------------------------------------------------------
__global__ void gemm_bias_relu(const float* __restrict__ X, const float* __restrict__ W,
                               const float* __restrict__ bias, float* __restrict__ Y,
                               int M, int K, int N, int ldY, int colOff, int doRelu) {
    int idx = blockIdx.x * 256 + threadIdx.x;
    if (idx >= M * N) return;
    int m = idx / N, n = idx - m * N;
    float acc = bias ? bias[n] : 0.f;
    const float* x = X + (size_t)m * K;
    for (int k = 0; k < K; ++k) acc += x[k] * W[(size_t)k * N + n];
    if (doRelu) acc = fmaxf(acc, 0.f);
    Y[(size_t)m * ldY + colOff + n] = acc;
}

// ---------------------------------------------------------------------------
// Row L2-normalize + convert to bf16 (one wave per row)
// ---------------------------------------------------------------------------
__global__ void rownorm_bf16(const float* __restrict__ X, __bf16* __restrict__ Y, int D) {
    int lane = threadIdx.x & 31;
    int row  = (blockIdx.x * 256 + threadIdx.x) >> 5;
    if (row >= BSZ) return;
    const float* x = X + (size_t)row * D;
    float s = 0.f;
    for (int i = lane; i < D; i += 32) s += x[i] * x[i];
    for (int off = 16; off; off >>= 1) s += __shfl_xor(s, off, 32);
    float inv = 1.0f / fmaxf(sqrtf(s), 1e-12f);
    for (int i = lane; i < D; i += 32) Y[(size_t)row * D + i] = f2bf(x[i] * inv);
}

// ---------------------------------------------------------------------------
// S = Nb @ Nb^T via v_wmma_f32_16x16x32_bf16 with TDM double-buffered LDS
// tiles. Fused epilogue: A0 = (i==j ? 0 : clamp(S,0,1)) + global sum for the
// mean threshold. Block tile 128x128, 8 waves (4 M x 2 N), 2x4 WMMA accs/wave.
// Wave 0 drives the Tensor Data Mover; TENSORcnt completes in order, so after
// issuing the next pair of tile loads, s_wait_tensorcnt<=2 guarantees the
// current pair has landed in LDS.
// ---------------------------------------------------------------------------
__global__ void __launch_bounds__(256)
simgemm_wmma(const __bf16* __restrict__ Nb, float* __restrict__ A0,
             int K, float* __restrict__ sumPtr) {
    __shared__ alignas(128) __bf16 at[2][128 * 32];
    __shared__ alignas(128) __bf16 bt[2][128 * 32];

    const int tid  = threadIdx.x;
    const int lane = tid & 31;
    const int wave = tid >> 5;      // 0..7
    const int wm   = wave & 3;      // 4 waves along M (32 rows each)
    const int wn   = wave >> 2;     // 2 waves along N (64 cols each)
    const int bm   = blockIdx.x & 31;
    const int bn   = blockIdx.x >> 5;
    const int rowBase = bm * 128;
    const int colBase = bn * 128;

    v8f acc[2][4];
#pragma unroll
    for (int i = 0; i < 2; ++i)
#pragma unroll
        for (int j = 0; j < 4; ++j)
#pragma unroll
            for (int r = 0; r < 8; ++r) acc[i][j][r] = 0.f;

    const unsigned atOff[2] = {(unsigned)(uintptr_t)&at[0][0],
                               (unsigned)(uintptr_t)&at[1][0]};
    const unsigned btOff[2] = {(unsigned)(uintptr_t)&bt[0][0],
                               (unsigned)(uintptr_t)&bt[1][0]};

    const int nch = K >> 5;         // K/32 chunks
    if (wave == 0) {
        tdm_load_tile(atOff[0], Nb, rowBase, K, 0);
        tdm_load_tile(btOff[0], Nb, colBase, K, 0);
    }

    for (int ch = 0; ch < nch; ++ch) {
        const int cur = ch & 1;
        if (wave == 0) {
            if (ch + 1 < nch) {
                tdm_load_tile(atOff[cur ^ 1], Nb, rowBase, K, (ch + 1) * 32);
                tdm_load_tile(btOff[cur ^ 1], Nb, colBase, K, (ch + 1) * 32);
                __builtin_amdgcn_s_wait_tensorcnt(2);   // current pair complete
            } else {
                __builtin_amdgcn_s_wait_tensorcnt(0);   // last pair complete
            }
        }
        __syncthreads();            // tile `cur` visible to all waves

        const int lr   = lane & 15;
        const int aoff = (lane < 16) ? 0 : 8;   // A frag: K = {o..o+7, o+16..o+23}
        const int boff = (lane < 16) ? 0 : 8;   // B frag: K = {o..o+7, o+16..o+23}

        v16bf bfr[4];
#pragma unroll
        for (int j = 0; j < 4; ++j) {
            const __bf16* p = &bt[cur][(wn * 64 + j * 16 + lr) * 32 + boff];
            v8bf blo = *(const v8bf*)p;
            v8bf bhi = *(const v8bf*)(p + 16);
            bfr[j] = __builtin_shufflevector(blo, bhi, 0, 1, 2, 3, 4, 5, 6, 7,
                                             8, 9, 10, 11, 12, 13, 14, 15);
        }
#pragma unroll
        for (int i = 0; i < 2; ++i) {
            const __bf16* p = &at[cur][(wm * 32 + i * 16 + lr) * 32];
            v8bf lo = *(const v8bf*)(p + aoff);
            v8bf hi = *(const v8bf*)(p + aoff + 16);
            v16bf afr = __builtin_shufflevector(lo, hi, 0, 1, 2, 3, 4, 5, 6, 7,
                                                8, 9, 10, 11, 12, 13, 14, 15);
#pragma unroll
            for (int j = 0; j < 4; ++j) {
                acc[i][j] = __builtin_amdgcn_wmma_f32_16x16x32_bf16(
                    false, afr, false, bfr[j], (short)0, acc[i][j], false, false);
            }
        }
        __syncthreads();            // reads done before TDM overwrites this buf
    }

    // Epilogue: diag-zero + clamp + store + partial sum.
    float psum = 0.f;
    const int rhalf = (lane >> 4) << 3;   // +8 rows for upper half-wave
    const int lcol  = lane & 15;
#pragma unroll
    for (int i = 0; i < 2; ++i) {
#pragma unroll
        for (int j = 0; j < 4; ++j) {
            int gcol = colBase + wn * 64 + j * 16 + lcol;
#pragma unroll
            for (int r = 0; r < 8; ++r) {
                int grow = rowBase + wm * 32 + i * 16 + rhalf + r;
                float s = acc[i][j][r];
                s = fminf(fmaxf(s, 0.f), 1.f);
                if (grow == gcol) s = 0.f;
                A0[(size_t)grow * BSZ + gcol] = s;
                psum += s;
            }
        }
    }
    for (int off = 16; off; off >>= 1) psum += __shfl_xor(psum, off, 32);
    if (lane == 0) atomicAdd(sumPtr, psum);
}

__global__ void mean_kernel(float* sumbuf) {
    sumbuf[1] = sumbuf[0] * (1.0f / ((float)BSZ * (float)BSZ));
}

// ---------------------------------------------------------------------------
// X0 init: rows < B/2 one-hot(labels), rest uniform 1/10
// ---------------------------------------------------------------------------
__global__ void init_X(float* __restrict__ X, const int* __restrict__ labels) {
    int idx = blockIdx.x * 256 + threadIdx.x;
    if (idx >= BSZ * NCLS) return;
    int b = idx / NCLS, c = idx - b * NCLS;
    X[idx] = (b < BSZ / 2) ? ((labels[b] == c) ? 1.0f : 0.0f) : (1.0f / NCLS);
}

// ---------------------------------------------------------------------------
// One GTG iteration. Effective A = (a < mean) ? 1 : 1-a (threshold + diversity
// fused). One wave per output row; X (160KB) staged transposed in dynamic LDS
// (CDNA5: 320KB/WGP). Lane 0 does the row-local update; entropy accumulated
// for the embedds graph. grid = 128 blocks x 256 threads, 32 rows/block.
// ---------------------------------------------------------------------------
__global__ void __launch_bounds__(256)
gtg_step(const float* __restrict__ A0, const float* __restrict__ sumbuf,
         const float* __restrict__ Xin, float* __restrict__ Xout,
         float* __restrict__ XsOut, float* __restrict__ entAcc, int t) {
    extern __shared__ float xlds[];          // [NCLS][BSZ], transposed
    const int tid = threadIdx.x;
    for (int idx = tid; idx < BSZ * NCLS; idx += 256) {
        int k = idx / NCLS, c = idx - k * NCLS;
        xlds[c * BSZ + k] = Xin[idx];
    }
    __syncthreads();

    const float mean = sumbuf[1];
    const int lane = tid & 31, wave = tid >> 5;

    for (int pass = 0; pass < 4; ++pass) {
        int row = blockIdx.x * 32 + pass * 8 + wave;
        const float* arow = A0 + (size_t)row * BSZ;
        float acc[NCLS];
#pragma unroll
        for (int c = 0; c < NCLS; ++c) acc[c] = 0.f;

        for (int k0 = lane * 4; k0 < BSZ; k0 += 128) {
            __builtin_prefetch(arow + k0 + 512, 0, 0);   // global_prefetch_b8
            float4 a4 = *(const float4*)(arow + k0);
            float a0 = (a4.x < mean) ? 1.f : 1.f - a4.x;
            float a1 = (a4.y < mean) ? 1.f : 1.f - a4.y;
            float a2 = (a4.z < mean) ? 1.f : 1.f - a4.z;
            float a3 = (a4.w < mean) ? 1.f : 1.f - a4.w;
#pragma unroll
            for (int c = 0; c < NCLS; ++c) {
                const float* xc = xlds + c * BSZ + k0;
                acc[c] += a0 * xc[0] + a1 * xc[1] + a2 * xc[2] + a3 * xc[3];
            }
        }
#pragma unroll
        for (int c = 0; c < NCLS; ++c)
            for (int off = 16; off; off >>= 1) acc[c] += __shfl_xor(acc[c], off, 32);

        if (lane == 0) {
            float x[NCLS], mult[NCLS], s = 0.f;
#pragma unroll
            for (int c = 0; c < NCLS; ++c) {
                x[c] = Xin[row * NCLS + c];
                mult[c] = x[c] * acc[c];
                s += mult[c];
            }
            float inv = 1.0f / (s + 1e-8f);
            float ent = 0.f;
#pragma unroll
            for (int c = 0; c < NCLS; ++c) {
                float d  = mult[c] * inv;
                float xn = x[c] + d;
                Xout[row * NCLS + c] = xn;
                if (XsOut) XsOut[(size_t)row * (NCLS * TITER) + c * TITER + t] = xn;
                ent -= d * logf(d + 1e-8f);
            }
            if (entAcc) entAcc[row] += ent * (1.0f / TITER);
        }
    }
}

// ---------------------------------------------------------------------------
// Final head + outputs
// ---------------------------------------------------------------------------
__global__ void ypred_kernel(const float* __restrict__ hcat, const float* __restrict__ w3,
                             const float* __restrict__ b3, float* __restrict__ out) {
    int b = blockIdx.x * 256 + threadIdx.x;
    if (b >= BSZ) return;
    float acc = b3[0];
    const float* h = hcat + (size_t)b * 300;
    for (int k = 0; k < 300; ++k) acc += h[k] * w3[k];
    out[b] = acc;
}

__global__ void finalize_kernel(const float* __restrict__ entAcc, float* __restrict__ out) {
    int b = blockIdx.x * 256 + threadIdx.x;
    if (b >= BSZ) return;
    out[BSZ + b]     = entAcc[b];
    out[2 * BSZ + b] = (b < BSZ / 2) ? 1.0f : 0.0f;
}

// ---------------------------------------------------------------------------
// Host orchestration
// ---------------------------------------------------------------------------
extern "C" void kernel_launch(void* const* d_in, const int* in_sizes, int n_in,
                              void* d_out, int out_size, void* d_ws, size_t ws_size,
                              hipStream_t stream) {
    (void)in_sizes; (void)n_in; (void)out_size; (void)ws_size;
    const float* feats[4] = {(const float*)d_in[0], (const float*)d_in[1],
                             (const float*)d_in[2], (const float*)d_in[3]};
    const float* embedds = (const float*)d_in[4];
    const int*   labels  = (const int*)d_in[6];
    const float* ls_w[4] = {(const float*)d_in[7],  (const float*)d_in[9],
                            (const float*)d_in[11], (const float*)d_in[13]};
    const float* ls_b[4] = {(const float*)d_in[8],  (const float*)d_in[10],
                            (const float*)d_in[12], (const float*)d_in[14]};
    const float* w1 = (const float*)d_in[15];
    const float* b1 = (const float*)d_in[16];
    const float* w2 = (const float*)d_in[17];
    const float* b2 = (const float*)d_in[18];
    const float* w3 = (const float*)d_in[19];
    const float* b3 = (const float*)d_in[20];
    float* out = (float*)d_out;

    char* ws = (char*)d_ws;
    auto alloc = [&](size_t bytes) {
        char* p = ws;
        ws += (bytes + 255) & ~(size_t)255;
        return p;
    };
    float*  A0     = (float*)alloc((size_t)BSZ * BSZ * 4);           // 64 MB
    __bf16* nb     = (__bf16*)alloc((size_t)BSZ * 512 * 2);          // 4 MB
    float*  pooled = (float*)alloc((size_t)BSZ * 512 * 4);           // 8 MB
    float*  emb    = (float*)alloc((size_t)BSZ * 128 * 4);           // 2 MB
    float*  Xa     = (float*)alloc((size_t)BSZ * NCLS * 4);
    float*  Xb     = (float*)alloc((size_t)BSZ * NCLS * 4);
    float*  xsbuf  = (float*)alloc((size_t)4 * BSZ * 300 * 4);       // 19.7 MB
    float*  h1     = (float*)alloc((size_t)BSZ * 150 * 4);
    float*  hcat   = (float*)alloc((size_t)BSZ * 300 * 4);
    float*  entAcc = (float*)alloc((size_t)BSZ * 4);
    float*  sumbuf = (float*)alloc(2 * sizeof(float));               // [sum, mean]

    const int CH[4] = {64, 128, 256, 512};
    const int S2[4] = {64, 64, 16, 16};
    const size_t dynLds = (size_t)BSZ * NCLS * sizeof(float);        // 160 KB

    for (int g = 0; g < 5; ++g) {
        const int Kdim = (g < 4) ? 128 : 512;
        if (g < 4) {
            const int C = CH[g];
            pool_kernel<<<(BSZ * C + 255) / 256, 256, 0, stream>>>(feats[g], pooled, C, S2[g]);
            gemm_bias_relu<<<(BSZ * 128 + 255) / 256, 256, 0, stream>>>(
                pooled, ls_w[g], ls_b[g], emb, BSZ, C, 128, 128, 0, 1);
            rownorm_bf16<<<BSZ * 32 / 256, 256, 0, stream>>>(emb, nb, 128);
        } else {
            rownorm_bf16<<<BSZ * 32 / 256, 256, 0, stream>>>(embedds, nb, 512);
        }

        hipMemsetAsync(sumbuf, 0, 2 * sizeof(float), stream);
        simgemm_wmma<<<1024, 256, 0, stream>>>(nb, A0, Kdim, sumbuf);
        mean_kernel<<<1, 1, 0, stream>>>(sumbuf);

        init_X<<<(BSZ * NCLS + 255) / 256, 256, 0, stream>>>(Xa, labels);
        if (g == 4) hipMemsetAsync(entAcc, 0, BSZ * sizeof(float), stream);

        float* Xc = Xa;
        float* Xn = Xb;
        float* xsOut = (g < 4) ? (xsbuf + (size_t)g * BSZ * 300) : nullptr;
        float* ea    = (g == 4) ? entAcc : nullptr;
        for (int t = 0; t < TITER; ++t) {
            gtg_step<<<128, 256, dynLds, stream>>>(A0, sumbuf, Xc, Xn, xsOut, ea, t);
            float* tmp = Xc; Xc = Xn; Xn = tmp;
        }

        if (g < 4) {
            gemm_bias_relu<<<(BSZ * 150 + 255) / 256, 256, 0, stream>>>(
                xsOut, w1, b1, h1, BSZ, 300, 150, 150, 0, 1);
            gemm_bias_relu<<<(BSZ * 75 + 255) / 256, 256, 0, stream>>>(
                h1, w2, b2, hcat, BSZ, 150, 75, 300, g * 75, 1);
        }
    }

    ypred_kernel<<<(BSZ + 255) / 256, 256, 0, stream>>>(hcat, w3, b3, out);
    finalize_kernel<<<(BSZ + 255) / 256, 256, 0, stream>>>(entAcc, out);
}
